// NGP_3040836846406
// MI455X (gfx1250) — matrix operand order repaired and checked by reading, not compiled, not run on hardware
//
#include <hip/hip_runtime.h>
#include <stdint.h>

typedef __attribute__((ext_vector_type(16))) _Float16 v16h;
typedef __attribute__((ext_vector_type(8)))  _Float16 v8h;
typedef __attribute__((ext_vector_type(8)))  float    v8f;
typedef __attribute__((ext_vector_type(4)))  float    v4f;

#define NLEVELS 16
#define TSIZE   (1u << 19)          // hashmap size per level (power of 2)

// single v_max_num_f32 (llvm.maxnum), not select+canonicalize
__device__ __forceinline__ float relu_f(float v) { return fmaxf(v, 0.0f); }

// One wave (32 lanes) processes a tile of 16 points.
//   lane = g*16 + r : r = point-in-tile (matrix column), g = half-wave group.
// WMMA orientation: D[M=neuron][N=point] = A(W^T tile, 16x32) x B(act^T, 32x16) + C
__global__ __launch_bounds__(256, 2) void ngp_fused_kernel(
    const float* __restrict__ xin,     // [N,3]
    const float* __restrict__ table,   // [16, 2^19, 2] -- 64 MB, L2-resident
    const float* __restrict__ w1,      // [32,64]
    const float* __restrict__ w2,      // [64,96]
    float* __restrict__ out,           // [N,96]
    int npts)
{
    __shared__ v16h sW1[4 * 32];        // w1^T (64x32) as 4 A-fragments   (4 KB)
    __shared__ v16h sW2[12 * 32];       // w2^T (96x64) as 6x2 A-fragments (12 KB)
    __shared__ _Float16 sH[8][16][64];  // [wave][point][neuron] layer-1 acts (16 KB)

    const int tid = threadIdx.x;

    // ---- cooperative fill: f32 weights -> f16 A-matrix fragments in LDS ----
    // A-layout (16x32 f16): lane group g holds K in {g*16..g*16+15},
    // VGPR v: K = g*16+2v+h ; row M = tile*16 + lane%16.
    for (int e = tid; e < 128 + 384; e += 256) {
        if (e < 128) {
            const int m = e >> 5, ln = e & 31, g = ln >> 4, n = m * 16 + (ln & 15);
            _Float16* p = (_Float16*)&sW1[e];
            #pragma unroll
            for (int v = 0; v < 8; ++v) {
                const int kb = g * 16 + 2 * v;
                p[2 * v + 0] = (_Float16)w1[(kb + 0) * 64 + n];   // w1^T[n][k] = w1[k*64+n]
                p[2 * v + 1] = (_Float16)w1[(kb + 1) * 64 + n];
            }
        } else {
            const int e2 = e - 128;
            const int ln = e2 & 31, g = ln >> 4;
            const int kc = (e2 >> 5) & 1, t = e2 >> 6;
            const int n = t * 16 + (ln & 15);
            _Float16* p = (_Float16*)&sW2[e2];
            #pragma unroll
            for (int v = 0; v < 8; ++v) {
                const int kb = kc * 32 + g * 16 + 2 * v;
                p[2 * v + 0] = (_Float16)w2[(kb + 0) * 96 + n];   // w2^T[n][k] = w2[k*96+n]
                p[2 * v + 1] = (_Float16)w2[(kb + 1) * 96 + n];
            }
        }
    }
    __syncthreads();

    const int lane = tid & 31;
    const int wid  = tid >> 5;
    const int g    = lane >> 4;
    const int r    = lane & 15;

    // layer-1 weight fragments stay in registers across the whole tile loop
    v16h a1[4];
    #pragma unroll
    for (int m = 0; m < 4; ++m) a1[m] = sW1[m * 32 + lane];

    const v8f zeroAcc = {0.f, 0.f, 0.f, 0.f, 0.f, 0.f, 0.f, 0.f};

    const int tiles = npts >> 4;               // 16 points per tile
    const int gwave = blockIdx.x * 8 + wid;
    const int nwave = gridDim.x * 8;

    for (int tile = gwave; tile < tiles; tile += nwave) {
        const int pt = tile * 16 + r;
        // x is streamed once: non-temporal load, keep L2 for the hash table
        const float x0 = __builtin_nontemporal_load(xin + pt * 3 + 0);
        const float x1 = __builtin_nontemporal_load(xin + pt * 3 + 1);
        const float x2 = __builtin_nontemporal_load(xin + pt * 3 + 2);
        const float xn0 = fminf(fmaxf(x0 * 0.5f + 0.5f, 0.f), 1.f);
        const float xn1 = fminf(fmaxf(x1 * 0.5f + 0.5f, 0.f), 1.f);
        const float xn2 = fminf(fmaxf(x2 * 0.5f + 0.5f, 0.f), 1.f);

        // ---- hash-grid encode: lane covers levels g*8 .. g*8+7 of its point ----
        // B-layout (32x16): lane holds column N=r, K = g*16 + e (e = vector elem)
        // K = level*2 + feat  =>  level = g*8 + j, benc[2j+f] = feature f.
        v16h benc;
        #pragma unroll
        for (int j = 0; j < 8; ++j) {
            const int lvl = g * 8 + j;
            // scale_l = 16 * 2^(8l/15) - 1 ; res = ceil(scale)+1 ; dense if res^3 <= T
            // (lvl is fixed per lane -> all of this is hoisted out of the tile loop)
            const float scale = 16.0f * exp2f((float)(lvl * 8) / 15.0f) - 1.0f;
            const uint32_t res = (uint32_t)ceilf(scale) + 1u;
            const bool dense =
                ((unsigned long long)res * res * res) <= (unsigned long long)TSIZE;
            const float px = xn0 * scale + 0.5f;
            const float py = xn1 * scale + 0.5f;
            const float pz = xn2 * scale + 0.5f;
            const float fx = floorf(px), fy = floorf(py), fz = floorf(pz);
            const float wx = px - fx, wy = py - fy, wz = pz - fz;
            const uint32_t ix = (uint32_t)fx, iy = (uint32_t)fy, iz = (uint32_t)fz;
            const float2* tb = (const float2*)table + (size_t)lvl * TSIZE;
            float f0 = 0.f, f1 = 0.f;
            #pragma unroll
            for (int c = 0; c < 8; ++c) {          // corner offsets: (bit2,bit1,bit0)
                const uint32_t ox = (c >> 2) & 1u, oy = (c >> 1) & 1u, oz = c & 1u;
                const uint32_t cx = ix + ox, cy = iy + oy, cz = iz + oz;
                const uint32_t hidx =
                    (cx ^ (cy * 2654435761u) ^ (cz * 805459861u)) & (TSIZE - 1u);
                const uint32_t didx = (cx + cy * res + cz * res * res) & (TSIZE - 1u);
                const uint32_t idx = dense ? didx : hidx;   // branchless (lanes diverge)
                const float wc = (ox ? wx : 1.f - wx) *
                                 (oy ? wy : 1.f - wy) *
                                 (oz ? wz : 1.f - wz);
                const float2 f = tb[idx];                   // global_load_b64 from L2
                f0 = fmaf(wc, f.x, f0);
                f1 = fmaf(wc, f.y, f1);
            }
            benc[2 * j + 0] = (_Float16)f0;
            benc[2 * j + 1] = (_Float16)f1;
        }

        // ---- layer 1: 4 x v_wmma_f32_16x16x32_f16, relu -> f16 -> LDS ----
        _Float16* hrow = &sH[wid][r][0];
        #pragma unroll
        for (int m = 0; m < 4; ++m) {
            v8f acc = __builtin_amdgcn_wmma_f32_16x16x32_f16(
                false, a1[m], false, benc, (short)0, zeroAcc, false, false);
            v8h pk;
            #pragma unroll
            for (int v = 0; v < 8; ++v) pk[v] = (_Float16)relu_f(acc[v]);
            // D: lane holds neurons m*16 + g*8 + v (8 consecutive) of point r
            *(v8h*)&hrow[m * 16 + g * 8] = pk;              // ds_store_b128
        }

        // same-wave LDS is in-order; barrier only stops compiler reordering
        asm volatile("" ::: "memory");

        // re-fragment h as layer-2 B operands: chunk kc -> K = kc*32 + g*16 + 0..15
        const v16h bh0 = *(const v16h*)&sH[wid][r][g * 16];
        const v16h bh1 = *(const v16h*)&sH[wid][r][32 + g * 16];

        // ---- layer 2: 6 tiles x 2 chained WMMAs, NT-coalesced b128 stores ----
        float* orow = out + (size_t)pt * 96 + g * 8;
        #pragma unroll
        for (int t = 0; t < 6; ++t) {
            const v16h a2lo = sW2[(2 * t + 0) * 32 + lane]; // ds_load_b128 x2
            const v16h a2hi = sW2[(2 * t + 1) * 32 + lane];
            v8f acc = __builtin_amdgcn_wmma_f32_16x16x32_f16(
                false, a2lo, false, bh0, (short)0, zeroAcc, false, false);
            acc = __builtin_amdgcn_wmma_f32_16x16x32_f16(
                false, a2hi, false, bh1, (short)0, acc, false, false);
            v4f o0, o1;
            o0.x = relu_f(acc[0]); o0.y = relu_f(acc[1]);
            o0.z = relu_f(acc[2]); o0.w = relu_f(acc[3]);
            o1.x = relu_f(acc[4]); o1.y = relu_f(acc[5]);
            o1.z = relu_f(acc[6]); o1.w = relu_f(acc[7]);
            // 96 MB output stream: non-temporal so the hash table stays in L2
            __builtin_nontemporal_store(o0, (v4f*)(orow + t * 16 + 0));
            __builtin_nontemporal_store(o1, (v4f*)(orow + t * 16 + 4));
        }
    }
}

extern "C" void kernel_launch(void* const* d_in, const int* in_sizes, int n_in,
                              void* d_out, int out_size, void* d_ws, size_t ws_size,
                              hipStream_t stream) {
    (void)n_in; (void)out_size; (void)d_ws; (void)ws_size;
    const float* x     = (const float*)d_in[0];
    const float* table = (const float*)d_in[1];
    const float* w1    = (const float*)d_in[2];
    const float* w2    = (const float*)d_in[3];
    float* out = (float*)d_out;

    const int npts  = in_sizes[0] / 3;
    const int tiles = npts / 16;
    int blocks = (tiles + 7) / 8;              // 8 waves per 256-thread block
    if (blocks > 1024) blocks = 1024;          // grid-stride over tiles
    if (blocks < 1) blocks = 1;
    ngp_fused_kernel<<<blocks, 256, 0, stream>>>(x, table, w1, w2, out, npts);
}
